// SinkhornOT_1099511628302
// MI455X (gfx1250) — matrix operand order repaired and checked by reading, not compiled, not run on hardware
//
#include <hip/hip_runtime.h>
#include <hip/hip_bf16.h>
#include <math.h>

typedef _Float16 v16h __attribute__((ext_vector_type(16)));
typedef _Float16 v8h  __attribute__((ext_vector_type(8)));
typedef float    v8f  __attribute__((ext_vector_type(8)));

#define DD 256
#define ROWS 64
#define EPS_INV 20.0f      /* 1/0.05 */
#define NUM_ITERS 15
#define MESH_ITERS 3
#define CP 33              /* padded stride for 32x32 LDS matrices */

__device__ __forceinline__ v16h cat8(v8h lo, v8h hi) {
  return __builtin_shufflevector(lo, hi, 0,1,2,3,4,5,6,7,8,9,10,11,12,13,14,15);
}

// A operand (16x32 f16) from a row-major [row][k] f16 buffer.
// 16-bit A layout: lane l holds row (l&15); K = klo..klo+7 and 16+klo..16+klo+7,
// klo = 0 (lanes 0-15) or 8 (lanes 16-31)  -> two contiguous 16B reads.
__device__ __forceinline__ v16h load_a(const _Float16* base, int stride,
                                       int row0, int k0, int lane) {
  int r   = row0 + (lane & 15);
  int klo = (lane < 16) ? 0 : 8;
  const _Float16* p = base + r * stride + k0 + klo;
  v8h lo = *(const v8h*)p;
  v8h hi = *(const v8h*)(p + 16);
  return cat8(lo, hi);
}

// B operand (32x16 f16) from a "B-transposed" row-major [n][k] f16 buffer.
// 16-bit B layout: lane l holds column (l&15); K = kb..kb+7 and kb+16..kb+23,
// kb = k0 (lanes 0-15) or k0+8 (lanes 16-31) -> two contiguous 16B reads.
__device__ __forceinline__ v16h load_bt(const _Float16* base, int stride,
                                        int n0, int k0, int lane) {
  int n  = n0 + (lane & 15);
  int kb = k0 + ((lane < 16) ? 0 : 8);
  const _Float16* p = base + n * stride + kb;
  v8h lo = *(const v8h*)p;
  v8h hi = *(const v8h*)(p + 16);
  return cat8(lo, hi);
}

__device__ __forceinline__ v8f wmma_f16(v16h a, v16h b, v8f c) {
  return __builtin_amdgcn_wmma_f32_16x16x32_f16(false, a, false, b,
                                                (short)0, c, false, false);
}

// One 64x256 MLP layer: dst = act(src @ W + bias), W given transposed [n][k] f16.
// k-outer loop: 4 A-tiles shared by both of this wave's column tiles (ct, ct+8),
// 8 live accumulators, 8 WMMAs per k-step. Unroll capped to keep VGPRs < 256.
__device__ __forceinline__ void mlp_layer(const _Float16* src, _Float16* dst,
                                          const _Float16* __restrict__ Wt,
                                          const float* __restrict__ bias,
                                          int lane, int wv, bool relu) {
  const int ct0 = wv;        // column tiles 0..7
  const int ct1 = wv + 8;    // column tiles 8..15
  const float bv0 = bias[ct0 * 16 + (lane & 15)];
  const float bv1 = bias[ct1 * 16 + (lane & 15)];
  v8f a00, a01, a02, a03, a10, a11, a12, a13;
  #pragma unroll
  for (int i = 0; i < 8; ++i) {
    a00[i] = bv0; a01[i] = bv0; a02[i] = bv0; a03[i] = bv0;
    a10[i] = bv1; a11[i] = bv1; a12[i] = bv1; a13[i] = bv1;
  }
  #pragma unroll 2
  for (int k8 = 0; k8 < 8; ++k8) {
    const int k0 = k8 * 32;
    v16h at0 = load_a(src, DD,  0, k0, lane);
    v16h at1 = load_a(src, DD, 16, k0, lane);
    v16h at2 = load_a(src, DD, 32, k0, lane);
    v16h at3 = load_a(src, DD, 48, k0, lane);
    v16h bm0 = load_bt(Wt, DD, ct0 * 16, k0, lane);
    v16h bm1 = load_bt(Wt, DD, ct1 * 16, k0, lane);
    a00 = wmma_f16(at0, bm0, a00);
    a01 = wmma_f16(at1, bm0, a01);
    a02 = wmma_f16(at2, bm0, a02);
    a03 = wmma_f16(at3, bm0, a03);
    a10 = wmma_f16(at0, bm1, a10);
    a11 = wmma_f16(at1, bm1, a11);
    a12 = wmma_f16(at2, bm1, a12);
    a13 = wmma_f16(at3, bm1, a13);
  }
  const int c0 = ct0 * 16 + (lane & 15);
  const int c1 = ct1 * 16 + (lane & 15);
  const int rb = (lane >= 16) ? 8 : 0;
  if (relu) {
    #pragma unroll
    for (int i = 0; i < 8; ++i) {
      dst[( 0 + rb + i) * DD + c0] = (_Float16)fmaxf(a00[i], 0.f);
      dst[(16 + rb + i) * DD + c0] = (_Float16)fmaxf(a01[i], 0.f);
      dst[(32 + rb + i) * DD + c0] = (_Float16)fmaxf(a02[i], 0.f);
      dst[(48 + rb + i) * DD + c0] = (_Float16)fmaxf(a03[i], 0.f);
      dst[( 0 + rb + i) * DD + c1] = (_Float16)fmaxf(a10[i], 0.f);
      dst[(16 + rb + i) * DD + c1] = (_Float16)fmaxf(a11[i], 0.f);
      dst[(32 + rb + i) * DD + c1] = (_Float16)fmaxf(a12[i], 0.f);
      dst[(48 + rb + i) * DD + c1] = (_Float16)fmaxf(a13[i], 0.f);
    }
  } else {
    #pragma unroll
    for (int i = 0; i < 8; ++i) {
      dst[( 0 + rb + i) * DD + c0] = (_Float16)a00[i];
      dst[(16 + rb + i) * DD + c0] = (_Float16)a01[i];
      dst[(32 + rb + i) * DD + c0] = (_Float16)a02[i];
      dst[(48 + rb + i) * DD + c0] = (_Float16)a03[i];
      dst[( 0 + rb + i) * DD + c1] = (_Float16)a10[i];
      dst[(16 + rb + i) * DD + c1] = (_Float16)a11[i];
      dst[(32 + rb + i) * DD + c1] = (_Float16)a12[i];
      dst[(48 + rb + i) * DD + c1] = (_Float16)a13[i];
    }
  }
}

// ---- prep: W[k][n] f32 -> Wt[n][k] f16 (both 256x256) ----
__global__ void prep_weights(const float* __restrict__ W1,
                             const float* __restrict__ W2,
                             _Float16* __restrict__ W1t,
                             _Float16* __restrict__ W2t) {
  int idx = blockIdx.x * blockDim.x + threadIdx.x;   // 0..131071
  int sel = idx >> 16;
  int e   = idx & 65535;
  int n = e >> 8, k = e & 255;
  const float* W = sel ? W2 : W1;
  _Float16*   Wt = sel ? W2t : W1t;
  Wt[n * DD + k] = (_Float16)W[k * DD + n];
}

__global__ void __launch_bounds__(256)
sinkhorn_kernel(const float* __restrict__ sq, const float* __restrict__ sr,
                const float* __restrict__ b1, const float* __restrict__ b2,
                const _Float16* __restrict__ W1t, const _Float16* __restrict__ W2t,
                float* __restrict__ out) {
  __shared__ _Float16 Xh[ROWS * DD];
  __shared__ _Float16 Hh[ROWS * DD];
  __shared__ _Float16 Yh[ROWS * DD];
  __shared__ float Cl[32 * CP];
  __shared__ float Kl[32 * CP];
  __shared__ float Tl[32 * CP];
  __shared__ float nrm[64];
  __shared__ float la[32], lb[32], rs[32], cs[32];
  __shared__ float costAcc;

  const int tid  = threadIdx.x;
  const int lane = tid & 31;
  const int wv   = tid >> 5;
  const int b    = blockIdx.x;

  if (tid == 0) {
    __builtin_prefetch(W1t, 0, 0);  // global_prefetch_b8 (weights resident in L2)
    __builtin_prefetch(W2t, 0, 0);
    costAcc = 0.f;
  }

  // stage X = [q ; r] as f16 in LDS (coalesced f32 reads)
  for (int i = tid; i < ROWS * DD; i += 256) {
    int row = i >> 8, k = i & 255;
    float v = (row < 32) ? sq[(size_t)b * 32 * DD + row * DD + k]
                         : sr[(size_t)b * 32 * DD + (row - 32) * DD + k];
    Xh[i] = (_Float16)v;
  }
  if (tid < 64) nrm[tid] = 0.f;
  __syncthreads();

  // ---- layer 1: H = relu(X @ W1 + b1) ----
  mlp_layer(Xh, Hh, W1t, b1, lane, wv, true);
  __syncthreads();

  // ---- layer 2: Y = H @ W2 + b2 ----
  mlp_layer(Hh, Yh, W2t, b2, lane, wv, false);
  __syncthreads();

  // ---- row squared-norms of Y (4 partial sums per row) ----
  {
    const int row = tid >> 2, part = tid & 3;
    float s = 0.f;
    for (int k = part * 64; k < part * 64 + 64; ++k) {
      float v = (float)Yh[row * DD + k];
      s += v * v;
    }
    atomicAdd(&nrm[row], s);
  }
  __syncthreads();

  // ---- C = cdist(q, r) via WMMA q.r^T : waves 0..3, one 16x16 tile each ----
  if (wv < 4) {
    const int tr = wv >> 1, tc = wv & 1;
    v8f acc;
    #pragma unroll
    for (int i = 0; i < 8; ++i) acc[i] = 0.f;
    #pragma unroll 2
    for (int k8 = 0; k8 < 8; ++k8) {
      const int k0 = k8 * 32;
      acc = wmma_f16(load_a(Yh, DD, tr * 16, k0, lane),
                     load_bt(Yh + 32 * DD, DD, tc * 16, k0, lane), acc);
    }
    const int m  = tc * 16 + (lane & 15);
    const int rb = (lane >= 16) ? 8 : 0;
    #pragma unroll
    for (int i = 0; i < 8; ++i) {
      const int k = tr * 16 + rb + i;
      float d2 = nrm[k] + nrm[32 + m] - 2.f * acc[i];
      float c  = sqrtf(fmaxf(d2, 0.f));
      Cl[k * CP + m] = c;
      Kl[k * CP + m] = -c * EPS_INV;
    }
  }
  if (tid < 32) { la[tid] = 0.f; lb[tid] = 0.f; }
  __syncthreads();

  // ---- Sinkhorn iterations (wave 0; in-wave LDS ops are ordered) ----
  if (wv == 0) {
    for (int it = 0; it < NUM_ITERS; ++it) {
      {
        const int k = lane;
        float mx = -3.0e38f;
        for (int m = 0; m < 32; ++m) mx = fmaxf(mx, Kl[k * CP + m] + lb[m]);
        float s = 0.f;
        for (int m = 0; m < 32; ++m) s += expf(Kl[k * CP + m] + lb[m] - mx);
        la[k] = -(mx + logf(s));
      }
      {
        const int m = lane;
        float mx = -3.0e38f;
        for (int k = 0; k < 32; ++k) mx = fmaxf(mx, Kl[k * CP + m] + la[k]);
        float s = 0.f;
        for (int k = 0; k < 32; ++k) s += expf(Kl[k * CP + m] + la[k] - mx);
        lb[m] = -(mx + logf(s));
      }
    }
  }
  __syncthreads();

  // T = exp(logK + log_a + log_b)
  for (int e = tid; e < 1024; e += 256) {
    int k = e >> 5, m = e & 31;
    Tl[k * CP + m] = expf(Kl[k * CP + m] + la[k] + lb[m]);
  }
  __syncthreads();

  // ---- mesh iterations ----
  for (int it = 0; it < MESH_ITERS; ++it) {
    for (int e = tid; e < 1024; e += 256) {
      int k = e >> 5, m = e & 31;
      float t = Tl[k * CP + m];
      Tl[k * CP + m] = t * t;
    }
    __syncthreads();
    if (tid < 32) {
      float s = 0.f;
      for (int m = 0; m < 32; ++m) s += Tl[tid * CP + m];
      rs[tid] = s + 1e-8f;
    }
    __syncthreads();
    for (int e = tid; e < 1024; e += 256) {
      int k = e >> 5, m = e & 31;
      Tl[k * CP + m] /= rs[k];
    }
    __syncthreads();
    if (tid < 32) {
      float s = 0.f;
      for (int k = 0; k < 32; ++k) s += Tl[k * CP + tid];
      cs[tid] = s + 1e-8f;
    }
    __syncthreads();
    for (int e = tid; e < 1024; e += 256) {
      int k = e >> 5, m = e & 31;
      Tl[k * CP + m] /= cs[m];
    }
    __syncthreads();
  }

  // ---- cost = sum(T*C); outputs ----
  {
    float part = 0.f;
    for (int e = tid; e < 1024; e += 256) {
      int k = e >> 5, m = e & 31;
      part += Tl[k * CP + m] * Cl[k * CP + m];
    }
    atomicAdd(&costAcc, part);
  }
  __syncthreads();

  const float cost = costAcc;
  float* simO  = out;
  float* TO    = out + 8192;
  float* CO    = out + 8192 + (size_t)8192 * 1024;
  float* costO = out + 8192 + (size_t)2 * 8192 * 1024;
  if (tid == 0) {
    simO[b]  = 1.f / (1.f + expf(cost));   // sigmoid(-cost)
    costO[b] = cost;
  }
  for (int e = tid; e < 1024; e += 256) {
    int k = e >> 5, m = e & 31;
    TO[(size_t)b * 1024 + e] = Tl[k * CP + m];
    CO[(size_t)b * 1024 + e] = Cl[k * CP + m];
  }
}

extern "C" void kernel_launch(void* const* d_in, const int* in_sizes, int n_in,
                              void* d_out, int out_size, void* d_ws, size_t ws_size,
                              hipStream_t stream) {
  const float* sq = (const float*)d_in[0];
  const float* sr = (const float*)d_in[1];
  const float* W1 = (const float*)d_in[2];
  const float* b1 = (const float*)d_in[3];
  const float* W2 = (const float*)d_in[4];
  const float* b2 = (const float*)d_in[5];
  _Float16* W1t = (_Float16*)d_ws;
  _Float16* W2t = W1t + 256 * 256;
  float* out = (float*)d_out;

  prep_weights<<<512, 256, 0, stream>>>(W1, W2, W1t, W2t);
  sinkhorn_kernel<<<8192, 256, 0, stream>>>(sq, sr, b1, b2, W1t, W2t, out);
}
